// InvariantPointAttentionMultimer_28621662060852
// MI455X (gfx1250) — matrix operand order repaired and checked by reading, not compiled, not run on hardware
//
#include <hip/hip_runtime.h>
#include <hip/hip_bf16.h>
#include <math.h>

// ---------------------------------------------------------------------------
// Types for the gfx1250 WMMA builtin: v_wmma_f32_16x16x32_bf16
// D(16x16 f32) = A(16x32 bf16) * B(32x16 bf16) + C
// ---------------------------------------------------------------------------
typedef __bf16 bf16_t;
typedef bf16_t v16bf __attribute__((ext_vector_type(16)));
typedef float  v8f   __attribute__((ext_vector_type(8)));

union BF16x16 { v16bf v; unsigned int d[8]; unsigned short h[16]; };
union F32x8   { v8f   v; float f[8]; };

__device__ __forceinline__ unsigned short f2bf(float f) {
  union { float f; unsigned u; } x; x.f = f;
  unsigned r = x.u + 0x7FFFu + ((x.u >> 16) & 1u);   // round-to-nearest-even
  return (unsigned short)(r >> 16);
}
__device__ __forceinline__ unsigned pack2(unsigned short lo, unsigned short hi) {
  return (unsigned)lo | ((unsigned)hi << 16);
}
// A-matrix (16x32 bf16) lane layout: lane m=lane&15; lanes<16 hold K base 0,
// lanes>=16 hold K base 8; VGPR j covers K = kb + (j<4 ? 2j : 8+2j), +1.
__device__ __forceinline__ int a_kofs(int lane, int j) {
  return ((lane < 16) ? 0 : 8) + ((j < 4) ? (2 * j) : (8 + 2 * j));
}
// B-matrix (32x16 bf16): lane n=lane&15; K base 0 (lanes<16) / 8; VGPR j: K=kb+(j<4 ? 2j : 8+2j), +1.
__device__ __forceinline__ int b_kofs(int lane, int j) {
  return ((lane < 16) ? 0 : 8) + ((j < 4) ? (2 * j) : (8 + 2 * j));
}
// C/D (16x16 f32): lane n=lane&15; VGPR j holds row m = j (+8 for lanes>=16).
__device__ __forceinline__ int cd_row(int lane, int j) {
  return j + ((lane < 16) ? 0 : 8);
}

// ---------------------------------------------------------------------------
// Problem constants
// ---------------------------------------------------------------------------
#define NN   768
#define CS   384
#define CZ   128
#define CH   16
#define HH   12
#define PQK  4
#define PV   8
#define FDIM 32            // extended feature width (16 qk + 12 pts + 2 rank1 + 2 pad)
#define VDIM 48            // v(16) + v_pts(24) + pad(8)
#define CATD 2112          // 192 + 4*96 + 1536  (== 66*32)
#define SQRT13 0.57735026919f
#define INFV 100000.0f

// ---------------------------------------------------------------------------
// Kernel 0: one-time transpose+convert of w_o -> bf16 [384][2112]
// ---------------------------------------------------------------------------
__global__ __launch_bounds__(256)
void prep_wo_kernel(const float* __restrict__ w_o, unsigned short* __restrict__ w_oT) {
  const long idx = (long)blockIdx.x * 256 + threadIdx.x;
  if (idx < (long)CS * CATD) {
    const int n = (int)(idx / CATD);
    const int kk = (int)(idx % CATD);
    w_oT[idx] = f2bf(w_o[(long)kk * CS + n]);
  }
}

// ---------------------------------------------------------------------------
// Kernel 1: projections + rigid transforms -> bf16 feature tensors
// qfeat/kfeat: [n][h][32];  vfeatT: [h][c(48)][n]  (K-contiguous for WMMA B)
// ---------------------------------------------------------------------------
__global__ __launch_bounds__(128)
void proj_kernel(const float* __restrict__ s, const float* __restrict__ rot,
                 const float* __restrict__ trans,
                 const float* __restrict__ w_q, const float* __restrict__ b_q,
                 const float* __restrict__ w_k, const float* __restrict__ b_k,
                 const float* __restrict__ w_v, const float* __restrict__ b_v,
                 const float* __restrict__ w_qp, const float* __restrict__ b_qp,
                 const float* __restrict__ w_kp, const float* __restrict__ b_kp,
                 const float* __restrict__ w_vp, const float* __restrict__ b_vp,
                 const float* __restrict__ head_weights,
                 unsigned short* __restrict__ qfeat,
                 unsigned short* __restrict__ kfeat,
                 unsigned short* __restrict__ vfeatT) {
  __shared__ float srow[CS];
  __shared__ float tmp[1152];
  const int n = blockIdx.x;
  const int tid = threadIdx.x;
  for (int i = tid; i < CS; i += blockDim.x) srow[i] = s[n * CS + i];
  __syncthreads();
  for (int t = tid; t < 1152; t += blockDim.x) {
    const float* w; const float* b; int c; int fo;
    if (t < 192)      { w = w_q;  b = b_q;  c = t;       fo = 192; }
    else if (t < 384) { w = w_k;  b = b_k;  c = t - 192; fo = 192; }
    else if (t < 576) { w = w_v;  b = b_v;  c = t - 384; fo = 192; }
    else if (t < 720) { w = w_qp; b = b_qp; c = t - 576; fo = 144; }
    else if (t < 864) { w = w_kp; b = b_kp; c = t - 720; fo = 144; }
    else              { w = w_vp; b = b_vp; c = t - 864; fo = 288; }
    float acc = b[c];
    for (int i = 0; i < CS; ++i) acc += srow[i] * w[i * fo + c];
    tmp[t] = acc;
  }
  __syncthreads();
  if (tid < HH) {
    const int h = tid;
    float R[9], T[3];
    for (int i = 0; i < 9; ++i) R[i] = rot[n * 9 + i];
    for (int i = 0; i < 3; ++i) T[i] = trans[n * 3 + i];
    const float hw = head_weights[h];
    const float sp = (hw > 20.f) ? hw : log1pf(expf(hw));
    const float pw = sqrtf(2.0f / (9.0f * (float)PQK)) * sp;
    unsigned short qf[FDIM], kf[FDIM], vf[VDIM];
    for (int c = 0; c < CH; ++c) {
      qf[c] = f2bf(tmp[h * CH + c] * 0.25f);   // * 1/sqrt(CH)
      kf[c] = f2bf(tmp[192 + h * CH + c]);
      vf[c] = f2bf(tmp[384 + h * CH + c]);
    }
    float q2 = 0.f, k2 = 0.f;
    for (int p = 0; p < PQK; ++p) {
      float ql[3], kl[3];
      for (int c = 0; c < 3; ++c) {
        ql[c] = tmp[576 + h * 12 + c * PQK + p];   // [H,3,P] layout
        kl[c] = tmp[720 + h * 12 + c * PQK + p];
      }
      for (int i = 0; i < 3; ++i) {
        float qg = R[i*3+0]*ql[0] + R[i*3+1]*ql[1] + R[i*3+2]*ql[2] + T[i];
        float kg = R[i*3+0]*kl[0] + R[i*3+1]*kl[1] + R[i*3+2]*kl[2] + T[i];
        q2 += qg * qg; k2 += kg * kg;
        qf[16 + p*3 + i] = f2bf(pw * qg);        // dot gives pw * (q.k)
        kf[16 + p*3 + i] = f2bf(kg);
      }
    }
    qf[28] = f2bf(-0.5f * pw * q2); kf[28] = f2bf(1.0f);  // -0.5*pw*q2 * 1
    qf[29] = f2bf(-0.5f * pw);      kf[29] = f2bf(k2);    // -0.5*pw   * k2
    qf[30] = 0; qf[31] = 0; kf[30] = 0; kf[31] = 0;
    for (int p = 0; p < PV; ++p) {
      float vl[3];
      for (int c = 0; c < 3; ++c) vl[c] = tmp[864 + h * 24 + c * PV + p];
      for (int i = 0; i < 3; ++i) {
        float vg = R[i*3+0]*vl[0] + R[i*3+1]*vl[1] + R[i*3+2]*vl[2] + T[i];
        vf[16 + p*3 + i] = f2bf(vg);
      }
    }
    for (int c = 40; c < VDIM; ++c) vf[c] = 0;
    unsigned short* qp = qfeat + ((long)n * HH + h) * FDIM;
    unsigned short* kp = kfeat + ((long)n * HH + h) * FDIM;
    for (int i = 0; i < FDIM; ++i) { qp[i] = qf[i]; kp[i] = kf[i]; }
    for (int c = 0; c < VDIM; ++c)
      vfeatT[((long)h * VDIM + c) * NN + n] = vf[c];
  }
}

// ---------------------------------------------------------------------------
// Kernel 2: fused pair-bias (z @ w_b, K=128) + per-head logits WMMA (K=32)
// one block per 16x16 (q,k) tile; z read #1 (cooperatively staged in LDS)
// ---------------------------------------------------------------------------
__global__ __launch_bounds__(256)
void logits_kernel(const float* __restrict__ z, const float* __restrict__ w_b,
                   const float* __restrict__ b_b,
                   const unsigned short* __restrict__ qfeat,
                   const unsigned short* __restrict__ kfeat,
                   float* __restrict__ logits) {
  __shared__ float bias_s[16][16][HH];        // [q_off][k_off][head]  12 KB
  __shared__ unsigned short wb_s[16][CZ];     // [head(pad16)][K]       4 KB
  __shared__ unsigned short za_s[256][32];    // [pair][K-window]      16 KB
  __shared__ float bb_s[16];
  const int q0 = blockIdx.x * 16, k0 = blockIdx.y * 16;
  const int tid = threadIdx.x;
  const int lane = tid & 31, wv = tid >> 5;
  const int m = lane & 15;

  // stage w_b (bf16, transposed, zero-padded to 16 heads) and b_b
  for (int t = tid; t < 16 * CZ; t += 256) {
    const int nh = t >> 7, kk = t & (CZ - 1);
    wb_s[nh][kk] = (nh < HH) ? f2bf(w_b[kk * HH + nh]) : (unsigned short)0;
  }
  if (tid < 16) bb_s[tid] = (tid < HH) ? b_b[tid] : 0.f;
  __syncthreads();

  // phase 1: bias tile. Each thread owns pair p=tid (qoff=tid>>4, koff=tid&15).
  const float* zrow = z + (((long)(q0 + (tid >> 4)) * NN) + (k0 + (tid & 15))) * CZ;
  F32x8 acc0, acc1;
  {
    const float binit = bb_s[m];
    for (int j = 0; j < 8; ++j) { acc0.f[j] = binit; acc1.f[j] = binit; }
  }
  const int t0 = wv, t1 = wv + 8;       // the two pair-subtiles this wave owns
  for (int ks = 0; ks < CZ; ks += 32) {
    __syncthreads();
    for (int c = 0; c < 32; ++c) za_s[tid][c] = f2bf(zrow[ks + c]);
    if (ks + 32 < CZ) __builtin_prefetch(zrow + ks + 32, 0, 3);
    __syncthreads();
    BF16x16 A0, A1, Bm;
    const unsigned int* r0 = (const unsigned int*)&za_s[t0 * 16 + m][0];
    const unsigned int* r1 = (const unsigned int*)&za_s[t1 * 16 + m][0];
    for (int j = 0; j < 8; ++j) {
      const int ka = a_kofs(lane, j) >> 1;
      A0.d[j] = r0[ka];
      A1.d[j] = r1[ka];
      Bm.d[j] = *(const unsigned int*)&wb_s[m][ks + b_kofs(lane, j)];
    }
    acc0.v = __builtin_amdgcn_wmma_f32_16x16x32_bf16(false, A0.v, false, Bm.v,
                                                     (short)0, acc0.v, false, false);
    acc1.v = __builtin_amdgcn_wmma_f32_16x16x32_bf16(false, A1.v, false, Bm.v,
                                                     (short)0, acc1.v, false, false);
  }
  for (int j = 0; j < 8; ++j) {
    const int mr = cd_row(lane, j);                 // k offset within subtile
    if (m < HH) {
      bias_s[t0][mr][m] = acc0.f[j];
      bias_s[t1][mr][m] = acc1.f[j];
    }
  }
  __syncthreads();

  // phase 2: per head, logits = qfeat . kfeat (K=32) + bias (C operand)
  for (int h = wv; h < HH; h += 8) {
    BF16x16 A, Bm; F32x8 acc;
    const int n = lane & 15;
    for (int j = 0; j < 8; ++j) acc.f[j] = bias_s[cd_row(lane, j)][n][h];
    const unsigned int* qf =
        (const unsigned int*)(qfeat + (((long)(q0 + m) * HH) + h) * FDIM);
    for (int j = 0; j < 8; ++j) A.d[j] = qf[a_kofs(lane, j) >> 1];
    const unsigned int* kf =
        (const unsigned int*)(kfeat + (((long)(k0 + n) * HH) + h) * FDIM);
    for (int j = 0; j < 8; ++j) Bm.d[j] = kf[b_kofs(lane, j) >> 1];
    acc.v = __builtin_amdgcn_wmma_f32_16x16x32_bf16(false, A.v, false, Bm.v,
                                                    (short)0, acc.v, false, false);
    for (int j = 0; j < 8; ++j) {
      const int mq = cd_row(lane, j);
      logits[((long)h * NN + (q0 + mq)) * NN + (k0 + n)] = acc.f[j];
    }
  }
}

// ---------------------------------------------------------------------------
// Kernel 3: masked softmax over k (axis -2), output bf16 attention
// ---------------------------------------------------------------------------
__global__ __launch_bounds__(256)
void softmax_kernel(const float* __restrict__ logits,
                    const float* __restrict__ mask,
                    unsigned short* __restrict__ abf) {
  __shared__ float red[256];
  __shared__ float rowbuf[NN];
  const int q = blockIdx.x, h = blockIdx.y, tid = threadIdx.x;
  const float* L = logits + ((long)h * NN + q) * NN;
  const float mq = mask[q];
  float mx = -1e30f;
  for (int k = tid; k < NN; k += 256) {
    const float x = (L[k] + INFV * (mq * mask[k] - 1.0f)) * SQRT13;
    rowbuf[k] = x;
    mx = fmaxf(mx, x);
  }
  red[tid] = mx; __syncthreads();
  for (int s = 128; s > 0; s >>= 1) {
    if (tid < s) red[tid] = fmaxf(red[tid], red[tid + s]);
    __syncthreads();
  }
  mx = red[0]; __syncthreads();
  float sum = 0.f;
  for (int k = tid; k < NN; k += 256) {
    const float e = expf(rowbuf[k] - mx);
    rowbuf[k] = e; sum += e;
  }
  red[tid] = sum; __syncthreads();
  for (int s = 128; s > 0; s >>= 1) {
    if (tid < s) red[tid] += red[tid + s];
    __syncthreads();
  }
  const float inv = 1.0f / red[0];
  unsigned short* out = abf + ((long)h * NN + q) * NN;
  for (int k = tid; k < NN; k += 256) out[k] = f2bf(rowbuf[k] * inv);
}

// ---------------------------------------------------------------------------
// Kernel 4: o / o_pt accumulation:  a[h] (768x768) @ vfeatT[h] (768x48)
// ---------------------------------------------------------------------------
__global__ __launch_bounds__(256)
void av_kernel(const unsigned short* __restrict__ abf,
               const unsigned short* __restrict__ vfeatT,
               float* __restrict__ oacc) {
  const int lane = threadIdx.x & 31;
  const int wid = blockIdx.x * 8 + (threadIdx.x >> 5);
  const int h = wid % HH;
  const int rem = wid / HH;
  const int ct = rem % 3, qt = rem / 3;
  const int q0 = qt * 16, c0 = ct * 16;
  const int m = lane & 15;
  F32x8 acc; for (int j = 0; j < 8; ++j) acc.f[j] = 0.f;
  const unsigned int* arow =
      (const unsigned int*)(abf + ((long)h * NN + q0 + m) * NN);
  const unsigned short* bcol = vfeatT + ((long)h * VDIM + c0 + m) * NN;
  for (int ks = 0; ks < NN; ks += 32) {
    BF16x16 A, Bm;
    for (int j = 0; j < 8; ++j) {
      A.d[j] = arow[(ks + a_kofs(lane, j)) >> 1];
      Bm.d[j] = *(const unsigned int*)(bcol + ks + b_kofs(lane, j));
    }
    acc.v = __builtin_amdgcn_wmma_f32_16x16x32_bf16(false, A.v, false, Bm.v,
                                                    (short)0, acc.v, false, false);
  }
  for (int j = 0; j < 8; ++j) {
    const int mq = cd_row(lane, j);
    oacc[((long)(q0 + mq) * HH + h) * VDIM + c0 + m] = acc.f[j];
  }
}

// ---------------------------------------------------------------------------
// Kernel 5: o_pair: per q, A = a[:,q,:] (12x768 pad 16) @ z[q] (768x128)
// z read #2 (cooperatively staged+transposed in LDS per 32-K tile)
// ---------------------------------------------------------------------------
__global__ __launch_bounds__(256)
void opair_kernel(const unsigned short* __restrict__ abf,
                  const float* __restrict__ z,
                  const unsigned short* __restrict__ zeropad,
                  float* __restrict__ opair) {
  __shared__ unsigned short zt_s[CZ][32];     // [c][k_local] 8 KB
  const int q = blockIdx.x;
  const int tid = threadIdx.x;
  const int lane = tid & 31;
  const int c0 = (tid >> 5) * 16;             // 8 waves -> 128 cols
  const int m = lane & 15;
  F32x8 acc; for (int j = 0; j < 8; ++j) acc.f[j] = 0.f;
  const unsigned int* arow = (const unsigned int*)
      ((m < HH) ? (abf + ((long)m * NN + q) * NN) : zeropad);
  const float* zq = z + (long)q * NN * CZ;
  // cooperative z-tile loader mapping: row r=tid>>3 (0..31), 16 cols each
  const int r = tid >> 3, cst = (tid & 7) * 16;
  for (int ks = 0; ks < NN; ks += 32) {
    __syncthreads();
    const float* src = zq + (long)(ks + r) * CZ + cst;
    for (int c = 0; c < 16; ++c) zt_s[cst + c][r] = f2bf(src[c]);
    if (ks + 32 < NN) __builtin_prefetch(zq + (long)(ks + 32 + r) * CZ + cst, 0, 3);
    __syncthreads();
    BF16x16 A, Bm;
    for (int j = 0; j < 8; ++j) {
      A.d[j] = arow[(ks + a_kofs(lane, j)) >> 1];
      Bm.d[j] = *(const unsigned int*)&zt_s[c0 + m][b_kofs(lane, j)];
    }
    acc.v = __builtin_amdgcn_wmma_f32_16x16x32_bf16(false, A.v, false, Bm.v,
                                                    (short)0, acc.v, false, false);
  }
  for (int j = 0; j < 8; ++j) {
    const int mh = cd_row(lane, j);
    if (mh < HH) opair[((long)q * HH + mh) * CZ + c0 + m] = acc.f[j];
  }
}

// ---------------------------------------------------------------------------
// Kernel 6: inverse rigid + norm + concat -> bf16 cat matrix [768 x 2112]
// ---------------------------------------------------------------------------
__global__ __launch_bounds__(256)
void cat_kernel(const float* __restrict__ oacc, const float* __restrict__ opair,
                const float* __restrict__ rot, const float* __restrict__ trans,
                unsigned short* __restrict__ catb) {
  const int q = blockIdx.x, tid = threadIdx.x;
  unsigned short* cq = catb + (long)q * CATD;
  for (int t = tid; t < 192; t += 256) {
    const int h = t >> 4, c = t & 15;
    cq[t] = f2bf(oacc[((long)q * HH + h) * VDIM + c]);
  }
  if (tid < 96) {                                   // j = h*PV + p
    const int h = tid >> 3, p = tid & 7;
    float g[3];
    for (int i = 0; i < 3; ++i)
      g[i] = oacc[((long)q * HH + h) * VDIM + 16 + p * 3 + i] - trans[q * 3 + i];
    float l[3];
    for (int i = 0; i < 3; ++i)                     // R^T (g - t)
      l[i] = rot[q*9 + 0*3 + i]*g[0] + rot[q*9 + 1*3 + i]*g[1] + rot[q*9 + 2*3 + i]*g[2];
    const float nr = sqrtf(l[0]*l[0] + l[1]*l[1] + l[2]*l[2] + 1e-8f);
    cq[192 + tid] = f2bf(l[0]);
    cq[288 + tid] = f2bf(l[1]);
    cq[384 + tid] = f2bf(l[2]);
    cq[480 + tid] = f2bf(nr);
  }
  for (int t = tid; t < HH * CZ; t += 256)
    cq[576 + t] = f2bf(opair[(long)q * HH * CZ + t]);
}

// ---------------------------------------------------------------------------
// Kernel 7: output GEMM [768 x 2112] @ w_oT (bf16) + b_o -> d_out (f32)
// ---------------------------------------------------------------------------
__global__ __launch_bounds__(256)
void out_kernel(const unsigned short* __restrict__ catb,
                const unsigned short* __restrict__ w_oT,
                const float* __restrict__ b_o,
                float* __restrict__ out) {
  const int lane = threadIdx.x & 31;
  const int wid = blockIdx.x * 8 + (threadIdx.x >> 5);
  const int nt = wid % 24, qt = wid / 24;           // 24 n-tiles, 48 q-tiles
  const int q0 = qt * 16, n0 = nt * 16;
  const int m = lane & 15;
  F32x8 acc;
  const float bn = b_o[n0 + m];
  for (int j = 0; j < 8; ++j) acc.f[j] = bn;
  const unsigned int* arow = (const unsigned int*)(catb + (long)(q0 + m) * CATD);
  const unsigned short* bcol = w_oT + (long)(n0 + m) * CATD;
  for (int ks = 0; ks < CATD; ks += 32) {
    BF16x16 A, Bm;
    for (int j = 0; j < 8; ++j) {
      A.d[j] = arow[(ks + a_kofs(lane, j)) >> 1];
      Bm.d[j] = *(const unsigned int*)(bcol + ks + b_kofs(lane, j));
    }
    acc.v = __builtin_amdgcn_wmma_f32_16x16x32_bf16(false, A.v, false, Bm.v,
                                                    (short)0, acc.v, false, false);
  }
  for (int j = 0; j < 8; ++j) {
    const int mq = cd_row(lane, j);
    out[(long)(q0 + mq) * CS + n0 + m] = acc.f[j];
  }
}

// ---------------------------------------------------------------------------
// Workspace layout (bytes)
// ---------------------------------------------------------------------------
#define OFF_QF    ((size_t)0)                               // 768*12*32*2
#define OFF_KF    (OFF_QF   + (size_t)NN*HH*FDIM*2)
#define OFF_VFT   (OFF_KF   + (size_t)NN*HH*FDIM*2)         // 12*48*768*2
#define OFF_LOG   (OFF_VFT  + (size_t)HH*VDIM*NN*2)         // 12*768*768*4
#define OFF_ABF   (OFF_LOG  + (size_t)HH*NN*NN*4)           // 12*768*768*2
#define OFF_OACC  (OFF_ABF  + (size_t)HH*NN*NN*2)           // 768*12*48*4
#define OFF_OPAIR (OFF_OACC + (size_t)NN*HH*VDIM*4)         // 768*12*128*4
#define OFF_CATB  (OFF_OPAIR+ (size_t)NN*HH*CZ*4)           // 768*2112*2
#define OFF_WOT   (OFF_CATB + (size_t)NN*CATD*2)            // 384*2112*2
#define OFF_ZERO  (OFF_WOT  + (size_t)CS*CATD*2)            // 2048 zero page

extern "C" void kernel_launch(void* const* d_in, const int* in_sizes, int n_in,
                              void* d_out, int out_size, void* d_ws, size_t ws_size,
                              hipStream_t stream) {
  (void)in_sizes; (void)n_in; (void)out_size; (void)ws_size;
  const float* s     = (const float*)d_in[0];
  const float* z     = (const float*)d_in[1];
  const float* rot   = (const float*)d_in[2];
  const float* trans = (const float*)d_in[3];
  const float* mask  = (const float*)d_in[4];
  const float* w_q   = (const float*)d_in[5];   const float* b_q = (const float*)d_in[6];
  const float* w_k   = (const float*)d_in[7];   const float* b_k = (const float*)d_in[8];
  const float* w_v   = (const float*)d_in[9];   const float* b_v = (const float*)d_in[10];
  const float* w_qp  = (const float*)d_in[11];  const float* b_qp = (const float*)d_in[12];
  const float* w_kp  = (const float*)d_in[13];  const float* b_kp = (const float*)d_in[14];
  const float* w_vp  = (const float*)d_in[15];  const float* b_vp = (const float*)d_in[16];
  const float* w_b   = (const float*)d_in[17];  const float* b_b  = (const float*)d_in[18];
  const float* hwts  = (const float*)d_in[19];
  const float* w_o   = (const float*)d_in[20];  const float* b_o  = (const float*)d_in[21];
  float* out = (float*)d_out;

  char* ws = (char*)d_ws;
  unsigned short* qfeat = (unsigned short*)(ws + OFF_QF);
  unsigned short* kfeat = (unsigned short*)(ws + OFF_KF);
  unsigned short* vfeatT= (unsigned short*)(ws + OFF_VFT);
  float*          logits= (float*)(ws + OFF_LOG);
  unsigned short* abf   = (unsigned short*)(ws + OFF_ABF);
  float*          oacc  = (float*)(ws + OFF_OACC);
  float*          opair = (float*)(ws + OFF_OPAIR);
  unsigned short* catb  = (unsigned short*)(ws + OFF_CATB);
  unsigned short* w_oT  = (unsigned short*)(ws + OFF_WOT);
  unsigned short* zpad  = (unsigned short*)(ws + OFF_ZERO);

  hipMemsetAsync(zpad, 0, 2048, stream);   // zero page for padded A rows
  prep_wo_kernel<<<((CS * CATD) + 255) / 256, 256, 0, stream>>>(w_o, w_oT);
  proj_kernel<<<NN, 128, 0, stream>>>(s, rot, trans, w_q, b_q, w_k, b_k, w_v, b_v,
                                      w_qp, b_qp, w_kp, b_kp, w_vp, b_vp, hwts,
                                      qfeat, kfeat, vfeatT);
  logits_kernel<<<dim3(NN/16, NN/16), 256, 0, stream>>>(z, w_b, b_b, qfeat, kfeat, logits);
  softmax_kernel<<<dim3(NN, HH), 256, 0, stream>>>(logits, mask, abf);
  av_kernel<<<(48 * 3 * HH) / 8, 256, 0, stream>>>(abf, vfeatT, oacc);
  opair_kernel<<<NN, 256, 0, stream>>>(abf, z, zpad, opair);
  cat_kernel<<<NN, 256, 0, stream>>>(oacc, opair, rot, trans, catb);
  out_kernel<<<(48 * 24) / 8, 256, 0, stream>>>(catb, w_oT, b_o, out);
}
